// MultiHeadAttention_53197464928286
// MI455X (gfx1250) — matrix operand order repaired and checked by reading, not compiled, not run on hardware
//
#include <hip/hip_runtime.h>
#include <hip/hip_bf16.h>

// ---------------------------------------------------------------------------
// MHA for MI455X (gfx1250, wave32).  Compute-bound (~142 GFLOP vs ~60MB I/O):
// everything routed through v_wmma_f32_16x16x32_f16 (f16 in, fp32 acc).
// Flash-attention structure: the 24x4096x4096 score tensor never touches HBM.
// WMMA operands are fed with ds_load_b128; strided operands are stored
// TRANSPOSED in LDS at staging time.  Pure-copy staging (Q, K tiles) uses
// gfx1250 async global->LDS (GLOBAL_LOAD_ASYNC_TO_LDS_B128 + s_wait_asynccnt).
// ---------------------------------------------------------------------------

typedef _Float16 half_t;
typedef __attribute__((ext_vector_type(4)))  _Float16 v4h;
typedef __attribute__((ext_vector_type(8)))  _Float16 v8h;
typedef __attribute__((ext_vector_type(16))) _Float16 v16h;
typedef __attribute__((ext_vector_type(8)))  float    v8f;

#define DEV_INLINE __device__ __forceinline__

// Problem constants
#define BATCH   2
#define SEQ     4096
#define DMODEL  768
#define HEADS   12
#define DK      64
#define ROWS    (BATCH * SEQ)   // 8192

// 1/sqrt(64) * log2(e): scores scaled into base-2 domain (v_exp_f32 is exp2)
#define SCORE_SCALE 0.18033688011112042f

// ---------------------------------------------------------------------------
// gfx1250 async global->LDS copy (CDNA5 ISA).
// LDS destination = low 32 bits of the generic pointer (aperture mapping
// truncates to addr[31:0] relative to the wave's LDS allocation).
// Tracked by ASYNCcnt; each thread waits its own count before the barrier.
// ---------------------------------------------------------------------------
DEV_INLINE void async_ld16(const half_t* lds_dst, const half_t* gsrc) {
  unsigned           loff = (unsigned)(unsigned long long)lds_dst;
  unsigned long long ga   = (unsigned long long)gsrc;
  asm volatile("global_load_async_to_lds_b128 %0, %1, off"
               :: "v"(loff), "v"(ga) : "memory");
}
DEV_INLINE void wait_async0() {
  asm volatile("s_wait_asynccnt 0" ::: "memory");
}

// ---------------------------------------------------------------------------
// WMMA fragment loaders (wave32 layouts, CDNA5 ISA §7.12.2).
// Each is two 16-byte LDS loads -> ds_load_b128.
// ---------------------------------------------------------------------------

DEV_INLINE v16h cat8(v8h lo, v8h hi) {
  return __builtin_shufflevector(lo, hi, 0, 1, 2, 3, 4, 5, 6, 7,
                                         8, 9, 10, 11, 12, 13, 14, 15);
}

// A-matrix 16x32 f16, element (m,k), source row-major, ld multiple of 8.
// lane 0..15  -> M = lane,     K = {0..7, 16..23}
// lane 16..31 -> M = lane-16,  K = {8..15, 24..31}
DEV_INLINE v16h load_a_frag(const half_t* __restrict__ base, int ld,
                            int row0, int k0, int lane) {
  const int m  = row0 + (lane & 15);
  const int kb = (lane < 16) ? 0 : 8;
  const half_t* p = base + m * ld + k0 + kb;
  return cat8(*(const v8h*)p, *(const v8h*)(p + 16));
}

// B-matrix 32x16 f16, operand element (k,n), SOURCE STORED TRANSPOSED as
// S[n][k] (contiguous along k), ld multiple of 8.
// lanes 0..15 carry K={0..7, 16..23}, lanes 16..31 carry K={8..15, 24..31}.
DEV_INLINE v16h load_b_frag_t(const half_t* __restrict__ base, int ld,
                              int k0, int n0, int lane) {
  const int n  = n0 + (lane & 15);
  const int kb = (lane < 16) ? 0 : 8;
  const half_t* p = base + n * ld + k0 + kb;
  return cat8(*(const v8h*)p, *(const v8h*)(p + 16));
}

DEV_INLINE v8f wmma_f16(v16h a, v16h b, v8f c) {
  return __builtin_amdgcn_wmma_f32_16x16x32_f16(
      /*neg_a=*/false, a, /*neg_b=*/false, b,
      /*c_mod=*/(short)0, c, /*reuse_a=*/false, /*reuse_b=*/false);
}

// ---------------------------------------------------------------------------
// GEMM: C[8192,768] = A[8192,768] @ W[768,768] + bias
// MODE 0: write f16 into per-head layout [B,H,T,64] (QKV projections)
// MODE 1: write f32 row-major [8192,768]            (output projection)
// Block 256 threads (8 waves).  Tile 128 rows x 64 cols, K staged 32-wide.
// (Staging converts fp32->f16, so it must pass through VGPRs: no async here.)
// ---------------------------------------------------------------------------
#define AS_LD 40   // 32 + 8 pad (halfs)
#define WT_LD 40   // W tile stored transposed: Wt[n (64)][k (32)] + pad

template <int MODE>
__global__ __launch_bounds__(256, 1)
void gemm768_kernel(const float* __restrict__ A,
                    const float* __restrict__ W,
                    const float* __restrict__ bias,
                    void* __restrict__ out) {
  __shared__ half_t As[128 * AS_LD];
  __shared__ half_t Wt[64 * WT_LD];

  const int tid  = threadIdx.x;
  const int lane = tid & 31;
  const int wave = tid >> 5;           // 0..7
  const int m0   = blockIdx.x * 128;   // row tile
  const int n0   = blockIdx.y * 64;    // col tile
  const int r0   = wave * 16;          // wave's row offset inside tile

  v8f acc[4] = {};                     // 16 rows x 64 cols per wave

  for (int k0 = 0; k0 < DMODEL; k0 += 32) {
    __syncthreads();
    {   // stage A tile 128x32 (fp32 -> f16)
      const int r  = tid >> 1;
      const int c0 = (tid & 1) * 16;
      const float4* src =
          reinterpret_cast<const float4*>(A + (size_t)(m0 + r) * DMODEL + k0 + c0);
      half_t* dst = &As[r * AS_LD + c0];
#pragma unroll
      for (int i = 0; i < 4; ++i) {
        float4 f = src[i];
        v4h hv = {(half_t)f.x, (half_t)f.y, (half_t)f.z, (half_t)f.w};
        *reinterpret_cast<v4h*>(dst + 4 * i) = hv;
      }
    }
    {   // stage W tile 32x64 TRANSPOSED -> Wt[n][k] (fp32 -> f16)
#pragma unroll
      for (int i = 0; i < 2; ++i) {
        const int idx = tid + 256 * i;       // 0..511
        const int r   = idx >> 4;            // k row 0..31
        const int c4  = (idx & 15) * 4;      // n col group
        float4 f = *reinterpret_cast<const float4*>(
            W + (size_t)(k0 + r) * DMODEL + n0 + c4);
        Wt[(c4 + 0) * WT_LD + r] = (half_t)f.x;
        Wt[(c4 + 1) * WT_LD + r] = (half_t)f.y;
        Wt[(c4 + 2) * WT_LD + r] = (half_t)f.z;
        Wt[(c4 + 3) * WT_LD + r] = (half_t)f.w;
      }
    }
    __syncthreads();

    const v16h a = load_a_frag(As, AS_LD, r0, 0, lane);
#pragma unroll
    for (int nt = 0; nt < 4; ++nt) {
      const v16h b = load_b_frag_t(Wt, WT_LD, 0, nt * 16, lane);
      acc[nt] = wmma_f16(a, b, acc[nt]);
    }
  }

  // Epilogue.  C/D layout: VGPR v, lanes 0..15 -> M=v, lanes 16..31 -> M=v+8.
  const int hi8 = (lane < 16) ? 0 : 8;
#pragma unroll
  for (int v = 0; v < 8; ++v) {
    const int mg = m0 + r0 + v + hi8;             // global row (b*T + t)
#pragma unroll
    for (int nt = 0; nt < 4; ++nt) {
      const int ng  = n0 + nt * 16 + (lane & 15); // global col (0..767)
      const float r = acc[nt][v] + bias[ng];
      if (MODE == 0) {
        const int b = mg / SEQ, t = mg % SEQ;
        const int h = ng / DK, d = ng % DK;
        ((half_t*)out)[(((size_t)b * HEADS + h) * SEQ + t) * DK + d] = (half_t)r;
      } else {
        ((float*)out)[(size_t)mg * DMODEL + ng] = r;
      }
    }
  }
}

// ---------------------------------------------------------------------------
// Flash attention: grid (T/128, B*H), block 256 (8 waves, 16 q-rows each).
// Per block: 128 queries x full d_k=64, streaming 64-key blocks.
// Q and K tiles staged with async global->LDS; V staged transposed through
// VGPRs (Vt[d][key]) so the P*V B-operand is contiguous.
// ---------------------------------------------------------------------------
#define QS_LD 72
#define KS_LD 72
#define VT_LD 72
#define PS_LD 72
#define SS_LD 68   // multiple of 4 -> aligned float4 softmax reads

__global__ __launch_bounds__(256, 1)
void flash_attn_kernel(const half_t* __restrict__ Qg,
                       const half_t* __restrict__ Kg,
                       const half_t* __restrict__ Vg,
                       float* __restrict__ attn) {
  __shared__ half_t Qs[128 * QS_LD];
  __shared__ half_t Ks[64 * KS_LD];
  __shared__ half_t Vt[64 * VT_LD];    // transposed: Vt[d][key]
  __shared__ half_t Ps[128 * PS_LD];
  __shared__ float  Ss[128 * SS_LD];
  __shared__ float  m_s[128], l_s[128], alpha_s[128];

  const int tid  = threadIdx.x;
  const int lane = tid & 31;
  const int wave = tid >> 5;              // 0..7
  const int q0   = blockIdx.x * 128;
  const int bh   = blockIdx.y;            // b*HEADS + h
  const int b    = bh / HEADS;
  const int h    = bh % HEADS;
  const int r0   = wave * 16;             // 16 query rows per wave
  const int hi8  = (lane < 16) ? 0 : 8;

  const half_t* qptr = Qg + (size_t)bh * SEQ * DK;
  const half_t* kptr = Kg + (size_t)bh * SEQ * DK;
  const half_t* vptr = Vg + (size_t)bh * SEQ * DK;

  {   // Q tile (128x64 f16): async copy, consumed after the in-loop wait+barrier
    const int r  = tid >> 1;
    const int c0 = (tid & 1) * 32;
    const half_t* src = qptr + (size_t)(q0 + r) * DK + c0;
    half_t* dst = &Qs[r * QS_LD + c0];
#pragma unroll
    for (int i = 0; i < 4; ++i) async_ld16(dst + 8 * i, src + 8 * i);
  }
  if (tid < 128) { m_s[tid] = -INFINITY; l_s[tid] = 0.0f; }

  v8f o[4] = {};   // 16 rows x 64 d-cols of O per wave (fp32)

  for (int j = 0; j < SEQ; j += 64) {
    __syncthreads();   // previous iteration fully consumed Ks/Vt/Ps

    {   // stage K async (row-major) and V through VGPRs (transposed)
      const int r  = tid >> 2;            // key 0..63
      const int c0 = (tid & 3) * 16;      // d offset {0,16,32,48}
      const half_t* ksrc = kptr + (size_t)(j + r) * DK + c0;
      half_t* kdst = &Ks[r * KS_LD + c0];
      async_ld16(kdst,     ksrc);
      async_ld16(kdst + 8, ksrc + 8);
      const v8h* vs = reinterpret_cast<const v8h*>(vptr + (size_t)(j + r) * DK + c0);
#pragma unroll
      for (int i = 0; i < 2; ++i) {
        v8h vv = vs[i];
#pragma unroll
        for (int e = 0; e < 8; ++e)
          Vt[(c0 + 8 * i + e) * VT_LD + r] = vv[e];
      }
    }
    wait_async0();     // own async copies (Q on iter 0, K every iter) landed
    __syncthreads();

    // S = Q * K^T (16 rows per wave x 64 key cols), scale into exp2 domain
    {
      v8f s[4] = {};
#pragma unroll
      for (int kc = 0; kc < 64; kc += 32) {
        const v16h a = load_a_frag(Qs, QS_LD, r0, kc, lane);
#pragma unroll
        for (int nt = 0; nt < 4; ++nt) {
          const v16h bk = load_b_frag_t(Ks, KS_LD, kc, nt * 16, lane);
          s[nt] = wmma_f16(a, bk, s[nt]);
        }
      }
#pragma unroll
      for (int nt = 0; nt < 4; ++nt)
#pragma unroll
        for (int v = 0; v < 8; ++v)
          Ss[(r0 + v + hi8) * SS_LD + nt * 16 + (lane & 15)] =
              s[nt][v] * SCORE_SCALE;
    }
    __syncthreads();

    // Online softmax in base-2: one thread per query row (threads 0..127).
    if (tid < 128) {
      const int r = tid;
      const float mprev = m_s[r];
      float mx = mprev;
      const float4* row = reinterpret_cast<const float4*>(&Ss[r * SS_LD]);
#pragma unroll
      for (int c = 0; c < 16; ++c) {
        float4 f = row[c];
        mx = fmaxf(mx, fmaxf(fmaxf(f.x, f.y), fmaxf(f.z, f.w)));
      }
      const float alpha = exp2f(mprev - mx);
      float sum = 0.0f;
      half_t* prow = &Ps[r * PS_LD];
#pragma unroll
      for (int c = 0; c < 16; ++c) {
        float4 f = row[c];
        float p0 = exp2f(f.x - mx), p1 = exp2f(f.y - mx);
        float p2 = exp2f(f.z - mx), p3 = exp2f(f.w - mx);
        v4h ph = {(half_t)p0, (half_t)p1, (half_t)p2, (half_t)p3};
        *reinterpret_cast<v4h*>(prow + 4 * c) = ph;
        sum += p0 + p1 + p2 + p3;
      }
      m_s[r] = mx;
      l_s[r] = l_s[r] * alpha + sum;
      alpha_s[r] = alpha;
    }
    __syncthreads();

    // Rescale O by alpha, then O += P * V.
    {
      float al[8];
#pragma unroll
      for (int v = 0; v < 8; ++v) al[v] = alpha_s[r0 + v + hi8];
#pragma unroll
      for (int nt = 0; nt < 4; ++nt)
#pragma unroll
        for (int v = 0; v < 8; ++v) o[nt][v] *= al[v];

#pragma unroll
      for (int kc = 0; kc < 64; kc += 32) {
        const v16h a = load_a_frag(Ps, PS_LD, r0, kc, lane);
#pragma unroll
        for (int nt = 0; nt < 4; ++nt) {
          const v16h bv = load_b_frag_t(Vt, VT_LD, kc, nt * 16, lane);
          o[nt] = wmma_f16(a, bv, o[nt]);
        }
      }
    }
  }

  __syncthreads();
  // Normalize by row sums; write fp32 into [B,T,DMODEL] (head-concat layout).
#pragma unroll
  for (int v = 0; v < 8; ++v) {
    const int rr  = r0 + v + hi8;
    const float inv = 1.0f / l_s[rr];
    const int qg  = q0 + rr;
#pragma unroll
    for (int nt = 0; nt < 4; ++nt) {
      attn[((size_t)b * SEQ + qg) * DMODEL + h * DK + nt * 16 + (lane & 15)] =
          o[nt][v] * inv;
    }
  }
}

// ---------------------------------------------------------------------------
// Host launcher
// ---------------------------------------------------------------------------
extern "C" void kernel_launch(void* const* d_in, const int* in_sizes, int n_in,
                              void* d_out, int out_size, void* d_ws, size_t ws_size,
                              hipStream_t stream) {
  (void)in_sizes; (void)n_in; (void)out_size; (void)ws_size;

  const float* x  = (const float*)d_in[0];
  const float* Wq = (const float*)d_in[1];
  const float* bq = (const float*)d_in[2];
  const float* Wk = (const float*)d_in[3];
  const float* bk = (const float*)d_in[4];
  const float* Wv = (const float*)d_in[5];
  const float* bv = (const float*)d_in[6];
  const float* Wo = (const float*)d_in[7];
  const float* bo = (const float*)d_in[8];

  char* ws = (char*)d_ws;
  const size_t head_elems = (size_t)BATCH * HEADS * SEQ * DK;   // 6.29M
  half_t* qh = (half_t*)ws;
  half_t* kh = (half_t*)(ws + head_elems * sizeof(half_t));
  half_t* vh = (half_t*)(ws + 2 * head_elems * sizeof(half_t));
  float*  at = (float*)(ws + 3 * head_elems * sizeof(half_t));  // [B,T,768] fp32

  dim3 gproj(ROWS / 128, DMODEL / 64);  // (64, 12)
  gemm768_kernel<0><<<gproj, 256, 0, stream>>>(x, Wq, bq, qh);
  gemm768_kernel<0><<<gproj, 256, 0, stream>>>(x, Wk, bk, kh);
  gemm768_kernel<0><<<gproj, 256, 0, stream>>>(x, Wv, bv, vh);

  dim3 gattn(SEQ / 128, BATCH * HEADS); // (32, 24)
  flash_attn_kernel<<<gattn, 256, 0, stream>>>(qh, kh, vh, at);

  gemm768_kernel<1><<<gproj, 256, 0, stream>>>(at, Wo, bo, d_out);
}
